// GNNLayer_50130858279044
// MI455X (gfx1250) — matrix operand order repaired and checked by reading, not compiled, not run on hardware
//
#include <hip/hip_runtime.h>
#include <hip/hip_bf16.h>
#include <math.h>

// ---------------------------------------------------------------------------
// TransformerConv (PyG) for MI455X / gfx1250, wave32 + WMMA.
//
// Phases:
//  1) pack_weights : Wq/Wk/Wv/Ws (f32) -> WMMA B-fragment bf16 hi/lo.
//  2) pack_x       : x (f32) -> WMMA A-fragment bf16 hi/lo (one pass,
//                    removes all cvt VALU from the 16x-reused GEMM loop).
//  3) qkvs_gemm    : q/k/v/skip GEMMs via v_wmma_f32_16x16x32_bf16 with the
//                    3-product hi/lo scheme (~fp32 accuracy). skip -> d_out.
//  4) init_stats   : node_max = -inf, node_sum = 0.
//  5) edge_alpha   : per-(edge,head) q·k/sqrt(C) + segment max (atomic).
//  6) edge_exp     : exp(alpha - max[dst]) + segment sum (atomicAdd).
//  7) edge_scatter : out[dst] += v[src] * (ex / denom)  (256 atomics/edge).
// ---------------------------------------------------------------------------

typedef __attribute__((ext_vector_type(16))) __bf16 v16bf;
typedef __attribute__((ext_vector_type(8)))  float  v8f;

#define N_NODES 50000
#define IN_CH   256
#define HEADS   8
#define CDIM    32
#define OUT_CH  256              // HEADS * CDIM
#define NEDGE   800000
#define MTILES  (N_NODES / 16)   // 3125, exact

// ---------------------------------------------------------------------------
// 1) Repack weights into WMMA B-fragment order, bf16 hi/lo split.
// B fragment (32x16 bf16, K x N): lane holds column n = lane&15,
// K-base = (lane>>4)*8, K = kbase + j (j<8), K = kbase + 8 + j (j>=8).
// Packed index: (((w*8 + kstep)*16 + ntile)*32 + lane)*16 + j
// ---------------------------------------------------------------------------
__global__ void pack_weights(const float* __restrict__ Wq, const float* __restrict__ Wk,
                             const float* __restrict__ Wv, const float* __restrict__ Ws,
                             __bf16* __restrict__ Bhi, __bf16* __restrict__ Blo) {
    int gid = blockIdx.x * blockDim.x + threadIdx.x;     // 4*8*16*32 = 16384
    if (gid >= 4 * 8 * 16 * 32) return;
    int lane  = gid & 31;
    int ntile = (gid >> 5) & 15;
    int kstep = (gid >> 9) & 7;
    int w     = gid >> 12;
    const float* W = (w == 0) ? Wq : (w == 1) ? Wk : (w == 2) ? Wv : Ws;
    int n  = ntile * 16 + (lane & 15);
    int kb = kstep * 32 + (lane >> 4) * 8;
    long base = (long)gid * 16;
    for (int j = 0; j < 16; ++j) {
        float f  = W[(kb + ((j < 8) ? j : (j + 8))) * OUT_CH + n];
        __bf16 h = (__bf16)f;
        Bhi[base + j] = h;
        Blo[base + j] = (__bf16)(f - (float)h);
    }
}

// ---------------------------------------------------------------------------
// 2) Repack x into WMMA A-fragment order, bf16 hi/lo split.
// A fragment (16x32 bf16): lane<16 -> M=lane,   K in {0..7, 16..23};
//                          lane>=16 -> M=lane-16, K in {8..15, 24..31}.
// Element j: K = kbase + j (j<8), K = kbase + 8 + j (j>=8), kbase=(lane>>4)*8.
// Packed index: ((mtile*8 + kstep)*32 + lane)*16 + j
// ---------------------------------------------------------------------------
__global__ void pack_x(const float* __restrict__ x,
                       __bf16* __restrict__ Ahi, __bf16* __restrict__ Alo) {
    int gid = blockIdx.x * blockDim.x + threadIdx.x;     // MTILES*8*32 = 800000
    if (gid >= MTILES * 8 * 32) return;
    int lane  = gid & 31;
    int kstep = (gid >> 5) & 7;
    int mtile = gid >> 8;
    int row = mtile * 16 + (lane & 15);
    int k0  = kstep * 32 + (lane >> 4) * 8;
    const float* xp = x + (long)row * IN_CH;
    long base = (long)gid * 16;
#pragma unroll
    for (int j = 0; j < 16; ++j) {
        int kk   = k0 + ((j < 8) ? j : (j + 8));
        float f  = xp[kk];
        __bf16 h = (__bf16)f;
        Ahi[base + j] = h;
        Alo[base + j] = (__bf16)(f - (float)h);
    }
}

// ---------------------------------------------------------------------------
// 3) Fused QKVS GEMM. grid = (3125 Mtiles, 4 weights, 4 n-quarters),
// block = 32 threads (one wave). Each wave: 16x64 output, K=256.
// Inner loop: pure fragment loads + 12 WMMAs per k-step, no conversion VALU.
// ---------------------------------------------------------------------------
__global__ void __launch_bounds__(32) qkvs_gemm(
    const __bf16* __restrict__ Ahi, const __bf16* __restrict__ Alo,
    const __bf16* __restrict__ Bhi, const __bf16* __restrict__ Blo,
    const float* __restrict__ bq, const float* __restrict__ bk,
    const float* __restrict__ bv, const float* __restrict__ bs,
    float* __restrict__ qo, float* __restrict__ ko,
    float* __restrict__ vo, float* __restrict__ so) {
    const int lane = threadIdx.x;
    const int mt   = blockIdx.x;   // M tile
    const int w    = blockIdx.y;   // weight select
    const int nq   = blockIdx.z;   // n quarter (4 tiles of 16)

    v8f acc[4] = {};

    for (int kstep = 0; kstep < 8; ++kstep) {
        const long aoff = (((long)mt * 8 + kstep) * 32 + lane) * 16;
        const v16bf ah = *(const v16bf*)(Ahi + aoff);
        const v16bf al = *(const v16bf*)(Alo + aoff);
#pragma unroll
        for (int t = 0; t < 4; ++t) {
            const int ntile = nq * 4 + t;
            const long boff = ((((long)w * 8 + kstep) * 16 + ntile) * 32 + lane) * 16;
            const v16bf bh = *(const v16bf*)(Bhi + boff);
            const v16bf bl = *(const v16bf*)(Blo + boff);
            // acc += Ahi*Bhi + Ahi*Blo + Alo*Bhi   (~fp32 accuracy)
            acc[t] = __builtin_amdgcn_wmma_f32_16x16x32_bf16(
                false, ah, false, bh, (short)0, acc[t], false, false);
            acc[t] = __builtin_amdgcn_wmma_f32_16x16x32_bf16(
                false, ah, false, bl, (short)0, acc[t], false, false);
            acc[t] = __builtin_amdgcn_wmma_f32_16x16x32_bf16(
                false, al, false, bh, (short)0, acc[t], false, false);
        }
    }

    const float* bias = (w == 0) ? bq : (w == 1) ? bk : (w == 2) ? bv : bs;
    float*       out  = (w == 0) ? qo : (w == 1) ? ko : (w == 2) ? vo : so;
    // C/D layout: col = n0 + (lane&15); VGPR r -> row m0 + r + 8*(lane>>4)
    const int rb = mt * 16 + (lane >> 4) * 8;
#pragma unroll
    for (int t = 0; t < 4; ++t) {
        const int col  = nq * 64 + t * 16 + (lane & 15);
        const float bc = bias[col];
#pragma unroll
        for (int r = 0; r < 8; ++r) {
            out[(long)(rb + r) * OUT_CH + col] = acc[t][r] + bc;
        }
    }
}

// ---------------------------------------------------------------------------
// 4) init per-(node,head) stats
// ---------------------------------------------------------------------------
__global__ void init_stats(float* __restrict__ node_max, float* __restrict__ node_sum, int n) {
    int i = blockIdx.x * 256 + threadIdx.x;
    if (i < n) {
        node_max[i] = -INFINITY;
        node_sum[i] = 0.0f;
    }
}

// Sign-split float atomic max (monotone under both int orderings).
__device__ inline void atomicMaxF(float* addr, float val) {
    if (val >= 0.0f) atomicMax((int*)addr, __float_as_int(val));
    else             atomicMin((unsigned int*)addr, __float_as_uint(val));
}

// ---------------------------------------------------------------------------
// 5) per-(edge,head) attention logit + segment max
// ---------------------------------------------------------------------------
__global__ void edge_alpha(const float* __restrict__ q, const float* __restrict__ k,
                           const int* __restrict__ src, const int* __restrict__ dst,
                           float* __restrict__ alpha, float* __restrict__ node_max) {
    int idx = blockIdx.x * 256 + threadIdx.x;
    if (idx >= NEDGE * HEADS) return;
    int e = idx >> 3, h = idx & 7;
    int s = src[e], d = dst[e];
    const float4* qp = (const float4*)(q + (long)d * OUT_CH + h * CDIM);
    const float4* kp = (const float4*)(k + (long)s * OUT_CH + h * CDIM);
    float acc = 0.0f;
#pragma unroll
    for (int j = 0; j < 8; ++j) {
        float4 a = qp[j], b = kp[j];
        acc += a.x * b.x + a.y * b.y + a.z * b.z + a.w * b.w;
    }
    acc *= 0.17677669529663687f;   // 1/sqrt(32)
    alpha[idx] = acc;
    atomicMaxF(&node_max[(long)d * HEADS + h], acc);
}

// ---------------------------------------------------------------------------
// 6) exp(alpha - max[dst]) + segment sum
// ---------------------------------------------------------------------------
__global__ void edge_exp(const int* __restrict__ dst, float* __restrict__ alpha,
                         const float* __restrict__ node_max, float* __restrict__ node_sum) {
    int idx = blockIdx.x * 256 + threadIdx.x;
    if (idx >= NEDGE * HEADS) return;
    int e = idx >> 3, h = idx & 7;
    int d = dst[e];
    float ex = expf(alpha[idx] - node_max[(long)d * HEADS + h]);
    alpha[idx] = ex;
    atomicAdd(&node_sum[(long)d * HEADS + h], ex);
}

// ---------------------------------------------------------------------------
// 7) scatter weighted messages: one 256-thread block per edge,
//    thread (h = tid>>5, c = tid&31) -> coalesced atomics per head row.
// ---------------------------------------------------------------------------
__global__ void edge_scatter(const int* __restrict__ src, const int* __restrict__ dst,
                             const float* __restrict__ v, const float* __restrict__ alpha,
                             const float* __restrict__ node_sum, float* __restrict__ out) {
    int e = blockIdx.x;
    int h = threadIdx.x >> 5;
    int c = threadIdx.x & 31;
    int s = src[e], d = dst[e];
    float attn = alpha[(long)e * HEADS + h] / node_sum[(long)d * HEADS + h];
    float val  = v[(long)s * OUT_CH + h * CDIM + c] * attn;
    atomicAdd(&out[(long)d * OUT_CH + h * CDIM + c], val);
}

// ---------------------------------------------------------------------------
extern "C" void kernel_launch(void* const* d_in, const int* in_sizes, int n_in,
                              void* d_out, int out_size, void* d_ws, size_t ws_size,
                              hipStream_t stream) {
    const float* x  = (const float*)d_in[0];
    const int*   ei = (const int*)d_in[1];
    const float* Wq = (const float*)d_in[2];
    const float* bq = (const float*)d_in[3];
    const float* Wk = (const float*)d_in[4];
    const float* bk = (const float*)d_in[5];
    const float* Wv = (const float*)d_in[6];
    const float* bv = (const float*)d_in[7];
    const float* Ws = (const float*)d_in[8];
    const float* bs = (const float*)d_in[9];
    const int* src = ei;             // edge_index[0]
    const int* dst = ei + NEDGE;     // edge_index[1]
    float* out = (float*)d_out;

    // workspace carve-up (256-B aligned)
    char* ws = (char*)d_ws;
    size_t off = 0;
    auto carve = [&](size_t bytes) -> void* {
        void* p = ws + off;
        off = (off + bytes + 255) & ~(size_t)255;
        return p;
    };
    float*  q        = (float*)carve((size_t)N_NODES * OUT_CH * sizeof(float));
    float*  k        = (float*)carve((size_t)N_NODES * OUT_CH * sizeof(float));
    float*  v        = (float*)carve((size_t)N_NODES * OUT_CH * sizeof(float));
    float*  alpha    = (float*)carve((size_t)NEDGE * HEADS * sizeof(float));
    float*  node_max = (float*)carve((size_t)N_NODES * HEADS * sizeof(float));
    float*  node_sum = (float*)carve((size_t)N_NODES * HEADS * sizeof(float));
    __bf16* Bhi      = (__bf16*)carve((size_t)4 * IN_CH * OUT_CH * sizeof(__bf16));
    __bf16* Blo      = (__bf16*)carve((size_t)4 * IN_CH * OUT_CH * sizeof(__bf16));
    __bf16* Ahi      = (__bf16*)carve((size_t)N_NODES * IN_CH * sizeof(__bf16));
    __bf16* Alo      = (__bf16*)carve((size_t)N_NODES * IN_CH * sizeof(__bf16));

    // 1) repack weights (tiny, once per launch)
    pack_weights<<<(16384 + 255) / 256, 256, 0, stream>>>(Wq, Wk, Wv, Ws, Bhi, Blo);

    // 2) repack x into A-fragment bf16 hi/lo (one streaming pass)
    const int nfrag = MTILES * 8 * 32;   // 800000
    pack_x<<<(nfrag + 255) / 256, 256, 0, stream>>>(x, Ahi, Alo);

    // 3) fused q/k/v/skip GEMMs (skip goes straight into d_out)
    qkvs_gemm<<<dim3(MTILES, 4, 4), 32, 0, stream>>>(Ahi, Alo, Bhi, Blo,
                                                     bq, bk, bv, bs, q, k, v, out);

    // 4) stats init
    const int nh = N_NODES * HEADS;
    init_stats<<<(nh + 255) / 256, 256, 0, stream>>>(node_max, node_sum, nh);

    // 5-6) logits + softmax statistics
    const int eh = NEDGE * HEADS;
    edge_alpha<<<(eh + 255) / 256, 256, 0, stream>>>(q, k, src, dst, alpha, node_max);
    edge_exp<<<(eh + 255) / 256, 256, 0, stream>>>(dst, alpha, node_max, node_sum);

    // 7) weighted scatter-add into d_out (already holds x@Ws + bs)
    edge_scatter<<<NEDGE, 256, 0, stream>>>(src, dst, v, alpha, node_sum, out);
}
